// CustomMultiHeadAttention_89970974917190
// MI455X (gfx1250) — compile-verified, matrix operand order checked by reading, not run on hardware
//
#include <hip/hip_runtime.h>

// ---------------------------------------------------------------------------
// Fused MHA for MI455X (gfx1250): bf16 WMMA (V_WMMA_F32_16X16X32_BF16)
//   K1: QKV = hs @ w_qkv^T + b  (fp32 in -> bf16 Q,K,V^T in ws)
//   K2: flash attention per (b, nh, 64-row q block), bf16 WMMA + online softmax
//   K3: out = attn @ w_out^T + b (bf16 A, fp32 out)
// ---------------------------------------------------------------------------

typedef __attribute__((ext_vector_type(16))) __bf16 v16bf;
typedef __attribute__((ext_vector_type(8)))  float  v8f;

namespace {
constexpr int B_  = 2;
constexpr int S_  = 2048;
constexpr int H_  = 2048;
constexpr int NH_ = 16;
constexpr int HD_ = 128;
constexpr size_t BSH_ = (size_t)B_ * S_ * H_;   // 8,388,608 elements
constexpr float SCALE_ = 0.08838834764831845f;  // 1/sqrt(128)

__device__ __forceinline__ unsigned short f2bf(float f) {
  union { float f; unsigned u; } c{f};
  unsigned r = c.u + 0x7FFFu + ((c.u >> 16) & 1u);  // round-to-nearest-even
  return (unsigned short)(r >> 16);
}
// order-preserving float -> uint key (for atomicMax-based float max)
__device__ __forceinline__ unsigned encf(float f) {
  union { float f; unsigned u; } c{f};
  return (c.u & 0x80000000u) ? ~c.u : (c.u | 0x80000000u);
}
__device__ __forceinline__ float decf(unsigned e) {
  union { unsigned u; float f; } c;
  c.u = (e & 0x80000000u) ? (e ^ 0x80000000u) : ~e;
  return c.f;
}

union FragU { v16bf v; uint4 q[2]; };

// Load one 16-bit A/B WMMA fragment: lane supplies two contiguous 16B runs
// (K = half*8..+8 and K = 16+half*8..+8) per the CDNA5 16-bit operand layout.
__device__ __forceinline__ v16bf ld_frag(const unsigned short* p) {
  FragU f;
  f.q[0] = *(const uint4*)(p);
  f.q[1] = *(const uint4*)(p + 16);
  return f.v;
}
}  // namespace

// ---------------------------------------------------------------------------
// GEMM: C[M,N] = A[M,K] * W[N,K]^T + bias.  Tile 128x128x32, 8 waves (4x2),
// each wave 32x64 = 2x4 WMMA accumulators.
// MODE 0: fp32 out row-major.  MODE 1: scatter bf16 into Q / K / V^T layout.
// TA = float (convert to bf16 while staging) or unsigned short (bf16 bits).
// ---------------------------------------------------------------------------
template <int MODE, typename TA>
__global__ __launch_bounds__(256) void mha_gemm_bf16wmma(
    const TA* __restrict__ A, const float* __restrict__ W,
    const float* __restrict__ bias, float* __restrict__ outF,
    unsigned short* __restrict__ qkvW, int M, int N, int K) {
  constexpr int LDT = 40;  // padded LDS row stride (bf16 elems), 80B = 16B-aligned
  __shared__ __align__(16) unsigned short As[128 * LDT];
  __shared__ __align__(16) unsigned short Bs[128 * LDT];

  const int t = threadIdx.x;
  const int mblock = blockIdx.y * 128, nblock = blockIdx.x * 128;
  const int row = t >> 1, seg = (t & 1) * 16;  // 2 threads per tile row

  const int lane = t & 31, half = lane >> 4, lr = lane & 15;
  const int waveId = t >> 5;
  const int wm = (waveId >> 1) * 32, wn = (waveId & 1) * 64;

  v8f acc[2][4];
#pragma unroll
  for (int mt = 0; mt < 2; ++mt)
#pragma unroll
    for (int nt = 0; nt < 4; ++nt)
#pragma unroll
      for (int i = 0; i < 8; ++i) acc[mt][nt][i] = 0.0f;

  for (int k0 = 0; k0 < K; k0 += 32) {
    __syncthreads();
    // ---- stage A tile (128 x 32) -> bf16 in LDS ----
    if constexpr (sizeof(TA) == 4) {
      const float* src = (const float*)A + (size_t)(mblock + row) * K + k0 + seg;
      float4 f0 = *(const float4*)(src + 0);
      float4 f1 = *(const float4*)(src + 4);
      float4 f2 = *(const float4*)(src + 8);
      float4 f3 = *(const float4*)(src + 12);
      uint4 lo = make_uint4(
          (unsigned)f2bf(f0.x) | ((unsigned)f2bf(f0.y) << 16),
          (unsigned)f2bf(f0.z) | ((unsigned)f2bf(f0.w) << 16),
          (unsigned)f2bf(f1.x) | ((unsigned)f2bf(f1.y) << 16),
          (unsigned)f2bf(f1.z) | ((unsigned)f2bf(f1.w) << 16));
      uint4 hi = make_uint4(
          (unsigned)f2bf(f2.x) | ((unsigned)f2bf(f2.y) << 16),
          (unsigned)f2bf(f2.z) | ((unsigned)f2bf(f2.w) << 16),
          (unsigned)f2bf(f3.x) | ((unsigned)f2bf(f3.y) << 16),
          (unsigned)f2bf(f3.z) | ((unsigned)f2bf(f3.w) << 16));
      *(uint4*)&As[row * LDT + seg] = lo;
      *(uint4*)&As[row * LDT + seg + 8] = hi;
      if (k0 + 32 < K) __builtin_prefetch(src + 32, 0, 1);
    } else {
      const unsigned short* src =
          (const unsigned short*)A + (size_t)(mblock + row) * K + k0 + seg;
      *(uint4*)&As[row * LDT + seg]     = *(const uint4*)(src);
      *(uint4*)&As[row * LDT + seg + 8] = *(const uint4*)(src + 8);
      if (k0 + 32 < K) __builtin_prefetch(src + 32, 0, 1);
    }
    // ---- stage W tile (128 rows of W, 32 k) -> bf16 in LDS (row n, k contiguous)
    {
      const float* src = W + (size_t)(nblock + row) * K + k0 + seg;
      float4 f0 = *(const float4*)(src + 0);
      float4 f1 = *(const float4*)(src + 4);
      float4 f2 = *(const float4*)(src + 8);
      float4 f3 = *(const float4*)(src + 12);
      uint4 lo = make_uint4(
          (unsigned)f2bf(f0.x) | ((unsigned)f2bf(f0.y) << 16),
          (unsigned)f2bf(f0.z) | ((unsigned)f2bf(f0.w) << 16),
          (unsigned)f2bf(f1.x) | ((unsigned)f2bf(f1.y) << 16),
          (unsigned)f2bf(f1.z) | ((unsigned)f2bf(f1.w) << 16));
      uint4 hi = make_uint4(
          (unsigned)f2bf(f2.x) | ((unsigned)f2bf(f2.y) << 16),
          (unsigned)f2bf(f2.z) | ((unsigned)f2bf(f2.w) << 16),
          (unsigned)f2bf(f3.x) | ((unsigned)f2bf(f3.y) << 16),
          (unsigned)f2bf(f3.z) | ((unsigned)f2bf(f3.w) << 16));
      *(uint4*)&Bs[row * LDT + seg] = lo;
      *(uint4*)&Bs[row * LDT + seg + 8] = hi;
      if (k0 + 32 < K) __builtin_prefetch(src + 32, 0, 1);
    }
    __syncthreads();

    v16bf af[2], bfr[4];
#pragma unroll
    for (int mt = 0; mt < 2; ++mt)
      af[mt] = ld_frag(&As[(wm + mt * 16 + lr) * LDT + half * 8]);
#pragma unroll
    for (int nt = 0; nt < 4; ++nt)
      bfr[nt] = ld_frag(&Bs[(wn + nt * 16 + lr) * LDT + half * 8]);
#pragma unroll
    for (int mt = 0; mt < 2; ++mt)
#pragma unroll
      for (int nt = 0; nt < 4; ++nt)
        acc[mt][nt] = __builtin_amdgcn_wmma_f32_16x16x32_bf16(
            false, af[mt], false, bfr[nt], (short)0, acc[mt][nt], false, false);
  }

  // ---- epilogue ----
#pragma unroll
  for (int mt = 0; mt < 2; ++mt)
#pragma unroll
    for (int nt = 0; nt < 4; ++nt) {
      const int ng = nblock + wn + nt * 16 + lr;     // output column
      const float bv = bias[ng];
      const int mg0 = mblock + wm + mt * 16 + half * 8;  // first output row
      if constexpr (MODE == 0) {
#pragma unroll
        for (int i = 0; i < 8; ++i)
          outF[(size_t)(mg0 + i) * N + ng] = acc[mt][nt][i] + bv;
      } else {
        const int m3 = ng >> 11, nh = (ng >> 7) & 15, hd = ng & 127;
#pragma unroll
        for (int i = 0; i < 8; ++i) {
          const int mg = mg0 + i;
          const int b = mg >> 11, s = mg & 2047;
          const unsigned short h = f2bf(acc[mt][nt][i] + bv);
          size_t idx;
          if (m3 == 2)  // V stored transposed: [b][nh][hd][s]
            idx = 2 * BSH_ + (((size_t)(b * NH_ + nh) * HD_ + hd) * S_ + s);
          else          // Q / K: [m3][b][nh][s][hd]
            idx = (size_t)m3 * BSH_ + (((size_t)(b * NH_ + nh) * S_ + s) * HD_ + hd);
          qkvW[idx] = h;
        }
      }
    }
}

// ---------------------------------------------------------------------------
// Flash attention. Grid: (S/64 q-blocks, B*NH). 256 threads = 8 waves (2M x 4N).
// Scores: 64x128 per kv block, K-dim = HD, Q/K frags straight from global (L2).
// P staged bf16 in LDS; O += P*V with V^T frags straight from global.
// Online softmax: shuffle reduce across 16-lane row groups + LDS atomics.
// ---------------------------------------------------------------------------
__global__ __launch_bounds__(256) void mha_flash_bf16wmma(
    const unsigned short* __restrict__ qkv, unsigned short* __restrict__ attnO) {
  const int t = threadIdx.x, lane = t & 31, half = lane >> 4, lr = lane & 15;
  const int waveId = t >> 5, waveM = waveId >> 2, waveN = waveId & 3;
  const int wm = waveM * 32, wn = waveN * 32;
  const int qbase = blockIdx.x * 64;
  const int bh = blockIdx.y, b = bh >> 4, nh = bh & 15;

  const unsigned short* qp = qkv + (size_t)bh * S_ * HD_;
  const unsigned short* kp = qkv + BSH_ + (size_t)bh * S_ * HD_;
  const unsigned short* vp = qkv + 2 * BSH_ + (size_t)bh * HD_ * S_;  // [hd][s]

  __shared__ __align__(16) unsigned short Pl[64 * 136];  // probs, padded stride
  __shared__ unsigned menc[64];  // running row max (order-preserving uint)
  __shared__ float mprev[64];    // running row max (float)
  __shared__ float lsum[64];     // running row sum
  __shared__ float rsc[64];      // per-row rescale factor this block

  if (t < 64) { mprev[t] = -3.0e38f; lsum[t] = 0.0f; }

  v8f o[2][2];
#pragma unroll
  for (int mt = 0; mt < 2; ++mt)
#pragma unroll
    for (int nt = 0; nt < 2; ++nt)
#pragma unroll
      for (int i = 0; i < 8; ++i) o[mt][nt][i] = 0.0f;

  for (int j = 0; j < S_; j += 128) {
    __syncthreads();                       // Pl / stats reusable
    if (t < 64) menc[t] = encf(mprev[t]);  // seed block max with running max
    __syncthreads();

    // ---- scores S = Q * K^T (this wave: 32 rows x 32 kv cols) ----
    v8f sc[2][2];
#pragma unroll
    for (int mt = 0; mt < 2; ++mt)
#pragma unroll
      for (int nt = 0; nt < 2; ++nt)
#pragma unroll
        for (int i = 0; i < 8; ++i) sc[mt][nt][i] = 0.0f;

#pragma unroll
    for (int kk = 0; kk < HD_; kk += 32) {
      v16bf aq[2], bk[2];
#pragma unroll
      for (int mt = 0; mt < 2; ++mt)
        aq[mt] = ld_frag(qp + (size_t)(qbase + wm + mt * 16 + lr) * HD_ + kk + half * 8);
#pragma unroll
      for (int nt = 0; nt < 2; ++nt)
        bk[nt] = ld_frag(kp + (size_t)(j + wn + nt * 16 + lr) * HD_ + kk + half * 8);
#pragma unroll
      for (int mt = 0; mt < 2; ++mt)
#pragma unroll
        for (int nt = 0; nt < 2; ++nt)
          sc[mt][nt] = __builtin_amdgcn_wmma_f32_16x16x32_bf16(
              false, aq[mt], false, bk[nt], (short)0, sc[mt][nt], false, false);
    }
    if (j + 128 < S_)  // prefetch next K block (L2 temporal)
      __builtin_prefetch(kp + (size_t)(j + 128 + (t >> 1)) * HD_ + (t & 1) * 64, 0, 1);

#pragma unroll
    for (int mt = 0; mt < 2; ++mt)
#pragma unroll
      for (int nt = 0; nt < 2; ++nt)
#pragma unroll
        for (int i = 0; i < 8; ++i) sc[mt][nt][i] *= SCALE_;

    // ---- block row max: shuffle over 16-lane row groups, then LDS atomicMax ----
#pragma unroll
    for (int mt = 0; mt < 2; ++mt) {
      float rm[8];
#pragma unroll
      for (int i = 0; i < 8; ++i) rm[i] = fmaxf(sc[mt][0][i], sc[mt][1][i]);
#pragma unroll
      for (int off = 1; off < 16; off <<= 1)
#pragma unroll
        for (int i = 0; i < 8; ++i) rm[i] = fmaxf(rm[i], __shfl_xor(rm[i], off, 32));
      if (lr == 0) {
        const int rb = wm + mt * 16 + half * 8;
#pragma unroll
        for (int i = 0; i < 8; ++i) atomicMax(&menc[rb + i], encf(rm[i]));
      }
    }
    __syncthreads();

    if (t < 64) {
      const float mn = decf(menc[t]);
      const float r = __expf(mprev[t] - mn);
      rsc[t] = r;
      lsum[t] *= r;
      mprev[t] = mn;  // mprev now holds m_new
    }
    __syncthreads();

    // ---- P = exp(S - m_new): store bf16 to LDS, accumulate row sums, rescale O
#pragma unroll
    for (int mt = 0; mt < 2; ++mt) {
      const int rb = wm + mt * 16 + half * 8;
      float mrow[8], rsum[8];
#pragma unroll
      for (int i = 0; i < 8; ++i) { mrow[i] = mprev[rb + i]; rsum[i] = 0.0f; }
#pragma unroll
      for (int nt = 0; nt < 2; ++nt) {
        const int col = wn + nt * 16 + lr;
#pragma unroll
        for (int i = 0; i < 8; ++i) {
          const float p = __expf(sc[mt][nt][i] - mrow[i]);
          Pl[(rb + i) * 136 + col] = f2bf(p);
          rsum[i] += p;
        }
      }
#pragma unroll
      for (int off = 1; off < 16; off <<= 1)
#pragma unroll
        for (int i = 0; i < 8; ++i) rsum[i] += __shfl_xor(rsum[i], off, 32);
      if (lr == 0)
#pragma unroll
        for (int i = 0; i < 8; ++i) atomicAdd(&lsum[rb + i], rsum[i]);
#pragma unroll
      for (int i = 0; i < 8; ++i) {
        const float r = rsc[rb + i];
        o[mt][0][i] *= r;
        o[mt][1][i] *= r;
      }
    }
    __syncthreads();

    // ---- O += P * V (K-dim = 128 kv; this wave: 32 rows x 32 hd cols) ----
#pragma unroll
    for (int kk = 0; kk < 128; kk += 32) {
      v16bf ap[2], bv[2];
#pragma unroll
      for (int mt = 0; mt < 2; ++mt)
        ap[mt] = ld_frag(&Pl[(wm + mt * 16 + lr) * 136 + kk + half * 8]);
#pragma unroll
      for (int nt = 0; nt < 2; ++nt)
        bv[nt] = ld_frag(vp + (size_t)(wn + nt * 16 + lr) * S_ + j + kk + half * 8);
#pragma unroll
      for (int mt = 0; mt < 2; ++mt)
#pragma unroll
        for (int nt = 0; nt < 2; ++nt)
          o[mt][nt] = __builtin_amdgcn_wmma_f32_16x16x32_bf16(
              false, ap[mt], false, bv[nt], (short)0, o[mt][nt], false, false);
    }
  }
  __syncthreads();

  // ---- epilogue: O / l, store bf16 to attn workspace [B,S,H] ----
#pragma unroll
  for (int mt = 0; mt < 2; ++mt) {
    const int rb = wm + mt * 16 + half * 8;
    float inv[8];
#pragma unroll
    for (int i = 0; i < 8; ++i) inv[i] = 1.0f / lsum[rb + i];
#pragma unroll
    for (int nt = 0; nt < 2; ++nt) {
      const int hd = wn + nt * 16 + lr;
#pragma unroll
      for (int i = 0; i < 8; ++i) {
        const int s = qbase + rb + i;
        attnO[((size_t)b * S_ + s) * H_ + nh * HD_ + hd] = f2bf(o[mt][nt][i] * inv[i]);
      }
    }
  }
}

// ---------------------------------------------------------------------------
extern "C" void kernel_launch(void* const* d_in, const int* in_sizes, int n_in,
                              void* d_out, int out_size, void* d_ws, size_t ws_size,
                              hipStream_t stream) {
  const float* hs    = (const float*)d_in[0];  // [B,S,H]
  const float* w_qkv = (const float*)d_in[1];  // [3H,H]
  const float* b_qkv = (const float*)d_in[2];  // [3H]
  const float* w_out = (const float*)d_in[3];  // [H,H]
  const float* b_out = (const float*)d_in[4];  // [H]
  float* out = (float*)d_out;                  // [B,S,H]

  unsigned short* qkv_ws  = (unsigned short*)d_ws;       // 3*BSH bf16 (48 MB)
  unsigned short* attn_ws = qkv_ws + 3 * BSH_;           // BSH bf16 (16 MB)

  const int M = B_ * S_;  // 4096
  dim3 blk(256);

  // 1) QKV projection -> Q, K, V^T (bf16) in workspace
  mha_gemm_bf16wmma<1, float><<<dim3((3 * H_) / 128, M / 128), blk, 0, stream>>>(
      hs, w_qkv, b_qkv, nullptr, qkv_ws, M, 3 * H_, H_);

  // 2) Flash attention -> attn output (bf16) in workspace
  mha_flash_bf16wmma<<<dim3(S_ / 64, B_ * NH_), blk, 0, stream>>>(qkv_ws, attn_ws);

  // 3) Output projection -> fp32 result
  mha_gemm_bf16wmma<0, unsigned short><<<dim3(H_ / 128, M / 128), blk, 0, stream>>>(
      attn_ws, w_out, b_out, out, nullptr, M, H_, H_);
}
